// PostProcessHOI_12352325943707
// MI455X (gfx1250) — compile-verified, matrix operand order checked by reading, not run on hardware
//
#include <hip/hip_runtime.h>
#include <cstdint>

// ---------------------------------------------------------------------------
// PostProcessHOI for MI455X (gfx1250, wave32).
//   B=1024 batches, Q=100 queries, C=81 obj classes, V=117 verb classes.
//   Kernel 1: softmax/argmax, sigmoid, box scale  -> d_out (vs0, scores,
//             labels, boxes).
//   Kernel 2: one 256-thread block per batch. Pairwise-NMS:
//             - kill relation precomputed as 100x100 bitmask (class-indep)
//             - per class step: rank column via V_WMMA_F32_16X16X32_F16
//               (Cmp matrix in A operand = documented layout; B = all-ones
//               so its layout is irrelevant; f32 accumulation of <=128 exact
//               ones -> exact integer ranks)
//             - greedy suppression: 4-word bit ops by thread 0
//             vs tile staged to LDS via global_load_async_to_lds_b128.
// ---------------------------------------------------------------------------

typedef __attribute__((ext_vector_type(16))) _Float16 v16h;
typedef __attribute__((ext_vector_type(8)))  float    v8f;

#define QN   100
#define CN   81
#define VN   117
#define QPAD 128

#if __has_builtin(__builtin_amdgcn_global_load_async_to_lds_b128)
#define HOI_HAVE_ASYNC 1
typedef int vi4 __attribute__((vector_size(16)));
typedef __attribute__((address_space(1))) vi4 gvi4;   // global int4
typedef __attribute__((address_space(3))) vi4 lvi4;   // LDS int4
#endif

__device__ __forceinline__ float iou4(const float* a, const float* c) {
  float a1 = (a[2] - a[0]) * (a[3] - a[1]);
  float a2 = (c[2] - c[0]) * (c[3] - c[1]);
  float lx = fmaxf(a[0], c[0]), ly = fmaxf(a[1], c[1]);
  float rx = fminf(a[2], c[2]), ry = fminf(a[3], c[3]);
  float w = fmaxf(rx - lx, 0.0f), h = fmaxf(ry - ly, 0.0f);
  float inter = w * h;
  return inter / (a1 + a2 - inter);
}

// ---------------------------------------------------------------------------
// Kernel 1: per (batch, query) prep.
// ---------------------------------------------------------------------------
__global__ void hoi_prep(const float* __restrict__ obj_logits,
                         const float* __restrict__ verb_logits,
                         const float* __restrict__ sub_boxes,
                         const float* __restrict__ obj_boxes,
                         const int*   __restrict__ tsizes,
                         float* __restrict__ out_vs,
                         float* __restrict__ out_sc,
                         float* __restrict__ out_lab,
                         float* __restrict__ out_box) {
  const int b = blockIdx.x;
  const int t = threadIdx.x;
  if (t >= QN) return;

  const float ih = (float)tsizes[b * 2 + 0];
  const float iw = (float)tsizes[b * 2 + 1];
  const long  bq = (long)b * QN + t;

  // softmax over 81 logits; max/argmax over first 80 probabilities
  const float* lg = obj_logits + bq * CN;
  float m = -3.4e38f;
  for (int c = 0; c < CN; ++c) m = fmaxf(m, lg[c]);
  float denom = 0.0f;
  for (int c = 0; c < CN; ++c) denom += __expf(lg[c] - m);
  float bm = lg[0]; int bi = 0;
  for (int c = 1; c < CN - 1; ++c) { float v = lg[c]; if (v > bm) { bm = v; bi = c; } }
  const float score = __expf(bm - m) / denom;

  out_sc[bq] = score;
  out_lab[(long)b * 2 * QN + t]      = 0.0f;        // SUBJECT_CATEGORY_ID
  out_lab[(long)b * 2 * QN + QN + t] = (float)bi;

  // boxes: cxcywh -> xyxy, scaled by (w,h,w,h)
  const float* sbp = sub_boxes + bq * 4;
  const float* obp = obj_boxes + bq * 4;
  float* sdst = out_box + ((long)b * 2 * QN + t) * 4;
  float* odst = out_box + ((long)b * 2 * QN + QN + t) * 4;
  sdst[0] = (sbp[0] - 0.5f * sbp[2]) * iw;
  sdst[1] = (sbp[1] - 0.5f * sbp[3]) * ih;
  sdst[2] = (sbp[0] + 0.5f * sbp[2]) * iw;
  sdst[3] = (sbp[1] + 0.5f * sbp[3]) * ih;
  odst[0] = (obp[0] - 0.5f * obp[2]) * iw;
  odst[1] = (obp[1] - 0.5f * obp[3]) * ih;
  odst[2] = (obp[0] + 0.5f * obp[2]) * iw;
  odst[3] = (obp[1] + 0.5f * obp[3]) * ih;

  // vs0 = sigmoid(verb_logits) * obj_score
  const float* vb  = verb_logits + bq * VN;
  float*       vso = out_vs + bq * VN;
  for (int v = 0; v < VN; ++v) vso[v] = score / (1.0f + __expf(-vb[v]));
}

// ---------------------------------------------------------------------------
// Kernel 2: pairwise-NMS, one block per batch.
// ---------------------------------------------------------------------------
__global__ void __launch_bounds__(256) hoi_pnms(float* __restrict__ out_vs,
                                                const float* __restrict__ out_lab,
                                                const float* __restrict__ out_box) {
  const int b    = blockIdx.x;
  const int tid  = threadIdx.x;
  const int lane = tid & 31;
  const int wave = tid >> 5;

  __shared__ __align__(16) float s_vs[QN * VN];   // 46800 B, pristine vs0
  __shared__ float    s_sb[QN][4];
  __shared__ float    s_ob[QN][4];
  __shared__ int      s_ol[QN];
  __shared__ unsigned s_km[QN][4];   // kill mask: bit q of row j
  __shared__ unsigned s_S[QN][4];    // supp of step i stored at row i
  __shared__ float    s_col[QPAD];
  __shared__ int      s_order[QN];

  float* vsb = out_vs + (long)b * QN * VN;

  // ---- stage vs tile into LDS (CDNA5 async path when available) ----------
#ifdef HOI_HAVE_ASYNC
  for (int idx = tid; idx < (QN * VN) / 4; idx += 256) {
    __builtin_amdgcn_global_load_async_to_lds_b128(
        (gvi4*)(vsb + 4 * idx), (lvi4*)(s_vs + 4 * idx), 0, 0);
  }
#else
  for (int idx = tid; idx < QN * VN; idx += 256) s_vs[idx] = vsb[idx];
#endif

  if (tid < 2 * QN) {
    const float* bp = out_box + ((long)b * 2 * QN + tid) * 4;
    float* dst = (tid < QN) ? &s_sb[tid][0] : &s_ob[tid - QN][0];
    dst[0] = bp[0]; dst[1] = bp[1]; dst[2] = bp[2]; dst[3] = bp[3];
  }
  if (tid < QN) s_ol[tid] = (int)out_lab[(long)b * 2 * QN + QN + tid];
  for (int t2 = tid; t2 < QN * 4; t2 += 256) ((unsigned*)s_S)[t2] = 0u;

#ifdef HOI_HAVE_ASYNC
#if __has_builtin(__builtin_amdgcn_s_wait_asynccnt)
  __builtin_amdgcn_s_wait_asynccnt(0);
#else
  asm volatile("s_wait_asynccnt 0" ::: "memory");
#endif
#endif
  __syncthreads();

  // ---- class-independent kill relation as 100x100 bitmask ----------------
  for (int idx = tid; idx < QN * 4; idx += 256) {
    const int j = idx >> 2, w = idx & 3;
    unsigned mword = 0u;
    for (int bb = 0; bb < 32; ++bb) {
      const int q = w * 32 + bb;
      if (q >= QN || q == j || s_ol[q] != s_ol[j]) continue;
      const float pi = fminf(iou4(&s_sb[j][0], &s_sb[q][0]),
                             iou4(&s_ob[j][0], &s_ob[q][0]));
      if (pi > 0.7f) mword |= (1u << bb);
    }
    s_km[j][w] = mword;
  }
  __syncthreads();

  // ---- WMMA rank setup: B operand = all ones (layout-agnostic) -----------
  v16h ones;
#pragma unroll
  for (int e = 0; e < 16; ++e) ones[e] = (_Float16)1.0f;

  const int m16   = lane & 15;   // A-matrix row within tile
  const int khalf = lane >> 4;   // K-half per documented 16-bit A 16x32 layout
  const int q     = wave * 16 + m16;

  // ---- 100 sequential class steps ----------------------------------------
  for (int i = 0; i < QN; ++i) {
    // (a) current column: vs0 minus accumulated suppression bit
    if (tid < QPAD) {
      float c = 0.0f;
      if (tid < QN)
        c = s_vs[tid * VN + i] - (float)((s_S[tid][i >> 5] >> (i & 31)) & 1u);
      s_col[tid] = c;
    }
    __syncthreads();

    // (b) rank[q] = sum_p cmp(p,q) via 4 chained v_wmma_f32_16x16x32_f16
    const float cq = s_col[q];
    v8f acc = {};
#pragma unroll
    for (int ch = 0; ch < 4; ++ch) {
      v16h amat;
#pragma unroll
      for (int e = 0; e < 16; ++e) {
        const int k = e + ((e & 8) ? 8 : 0) + (khalf << 3);  // documented A map
        const int p = ch * 32 + k;
        const float cp = s_col[p];
        const bool hit = (p < QN) && ((cp > cq) || (cp == cq && p < q));
        amat[e] = hit ? (_Float16)1.0f : (_Float16)0.0f;
      }
      acc = __builtin_amdgcn_wmma_f32_16x16x32_f16(
          false, amat, false, ones, (short)0, acc, false, false);
    }
    // extract via documented C/D layout: lane0 rows 0-7, lane16 rows 8-15
    if (m16 == 0) {
      const int qbase = wave * 16 + (khalf ? 8 : 0);
#pragma unroll
      for (int r = 0; r < 8; ++r) {
        const int qq = qbase + r;
        if (qq < QN) s_order[(int)(acc[r] + 0.5f)] = qq;
      }
    }
    __syncthreads();

    // (c) greedy suppression over bitmasks, thread 0
    if (tid == 0) {
      unsigned alive[4] = {~0u, ~0u, ~0u, 0xFu};
      unsigned supp[4]  = {0u, 0u, 0u, 0u};
      for (int r = 0; r < QN; ++r) {
        const int j = s_order[r];
        const int jw = j >> 5;
        const unsigned jb = 1u << (j & 31);
        if (alive[jw] & jb) {
#pragma unroll
          for (int w2 = 0; w2 < 4; ++w2) {
            const unsigned k = s_km[j][w2] & alive[w2];
            alive[w2] &= ~k;
            supp[w2]  |= k;
          }
          alive[jw] &= ~jb;
        }
      }
      s_S[i][0] = supp[0]; s_S[i][1] = supp[1];
      s_S[i][2] = supp[2]; s_S[i][3] = supp[3];
    }
    __syncthreads();
  }

  // ---- write vs_out = vs0 - S[row][col] (matches JAX row-scatter) --------
  for (int idx = tid; idx < QN * VN; idx += 256) {
    const int qq = idx / VN;
    const int cc = idx - qq * VN;
    float v = s_vs[idx];
    if (cc < QN) v -= (float)((s_S[qq][cc >> 5] >> (cc & 31)) & 1u);
    vsb[idx] = v;
  }
}

// ---------------------------------------------------------------------------
extern "C" void kernel_launch(void* const* d_in, const int* in_sizes, int n_in,
                              void* d_out, int out_size, void* d_ws, size_t ws_size,
                              hipStream_t stream) {
  const float* obj_logits  = (const float*)d_in[0];
  const float* verb_logits = (const float*)d_in[1];
  const float* sub_boxes   = (const float*)d_in[2];
  const float* obj_boxes   = (const float*)d_in[3];
  const int*   tsizes      = (const int*)d_in[4];

  const int B = in_sizes[4] / 2;   // target_sizes is (B,2)

  float* out     = (float*)d_out;
  float* out_vs  = out;                               // B*Q*V
  float* out_sc  = out_vs  + (size_t)B * QN * VN;     // B*Q
  float* out_lab = out_sc  + (size_t)B * QN;          // B*2Q
  float* out_box = out_lab + (size_t)B * 2 * QN;      // B*2Q*4

  hipLaunchKernelGGL(hoi_prep, dim3(B), dim3(128), 0, stream,
                     obj_logits, verb_logits, sub_boxes, obj_boxes, tsizes,
                     out_vs, out_sc, out_lab, out_box);
  hipLaunchKernelGGL(hoi_pnms, dim3(B), dim3(256), 0, stream,
                     out_vs, out_lab, out_box);
}
